// MultiheadAttention_53102975648112
// MI455X (gfx1250) — matrix operand order repaired and checked by reading, not compiled, not run on hardware
//
#include <hip/hip_runtime.h>
#include <hip/hip_bf16.h>

// ---------------------------------------------------------------------------
// CDNA5 (gfx1250) fused multi-head attention, bf16 WMMA pipeline.
//   B=2, S=2048, E=1024, H=16, K=64
// Stages:
//   1) convert x -> bf16 ; wq/wk/wv -> transposed bf16 (N x E) ; wo -> bf16
//   2) WMMA GEMM (A * Bt^T), 32x64 register blocking per wave:
//      Q, K, V (V stored transposed [b,h,d,s])
//   3) flash attention (causal, online softmax) -> O bf16
//   4) WMMA GEMM: out = O * Wo^T  (f32 output)
// ---------------------------------------------------------------------------

#define S_LEN   2048
#define N_HEADS 16
#define HD      64
#define EMB     1024   // = N_HEADS * HD = D_MODEL
#define MROWS   4096   // B * S

typedef __attribute__((ext_vector_type(16))) __bf16 v16bf;
typedef __attribute__((ext_vector_type(8)))  float  v8f;

struct alignas(16) U128 { unsigned int x, y, z, w; };

union Frag16 {
  v16bf v;
  U128  q[2];
  unsigned short s[16];
};

__device__ __forceinline__ unsigned short f2bf(float f) {
  unsigned int u = __float_as_uint(f);
  u += 0x7FFFu + ((u >> 16) & 1u);   // round to nearest even
  return (unsigned short)(u >> 16);
}

// A-matrix fragment (16x32 bf16, M x K): lane = row (lane%16).
// lanes 0-15: K = 0..15; lanes 16-31: K = 16..31.
__device__ __forceinline__ v16bf load_afrag(const unsigned short* rowPtr, int lane) {
  const int off = (lane < 16) ? 0 : 16;
  Frag16 f;
  f.q[0] = *(const U128*)(rowPtr + off);
  f.q[1] = *(const U128*)(rowPtr + off + 8);
  return f.v;
}

// B-matrix fragment (32x16 bf16, K x N) from B^T stored row-major (N x Kd):
// lane = column (lane%16); lanes 0-15 hold K=0..15, lanes 16-31 hold K=16..31.
__device__ __forceinline__ v16bf load_bfrag(const unsigned short* colPtr, int lane) {
  const int off = (lane < 16) ? 0 : 16;
  Frag16 f;
  f.q[0] = *(const U128*)(colPtr + off);
  f.q[1] = *(const U128*)(colPtr + off + 8);
  return f.v;
}

__device__ __forceinline__ v8f wmma_bf16(v16bf a, v16bf b, v8f c) {
  // (neg_a, A, neg_b, B, c_mod, C, reuse_a, reuse_b)
  return __builtin_amdgcn_wmma_f32_16x16x32_bf16(false, a, false, b, (short)0, c,
                                                 false, false);
}

// ---------------------------------------------------------------------------
// Conversion kernels
// ---------------------------------------------------------------------------
__global__ void convert_bf16_kernel(const float* __restrict__ x,
                                    unsigned short* __restrict__ y, int n) {
  int i = blockIdx.x * blockDim.x + threadIdx.x;
  if (i < n) y[i] = f2bf(x[i]);
}

// W is (E x 1024) row-major; write Wt (1024 x E) row-major (bf16).
__global__ void transpose_convert_kernel(const float* __restrict__ w,
                                         unsigned short* __restrict__ wt) {
  int i = blockIdx.x * blockDim.x + threadIdx.x;   // 0 .. 1M-1
  int e = i >> 10;
  int n = i & 1023;
  wt[(size_t)n * EMB + e] = f2bf(w[i]);
}

// ---------------------------------------------------------------------------
// GEMM: C[M x N] = A[M x Kd] * Bt[N x Kd]^T   (bf16 in, f32 accumulate)
// Block = 128 threads (4 waves); each wave -> 32x64 strip of C
// (two 16-row A strips sharing the 4 B fragments: 8 WMMA / 12 loads per
//  32-wide k-step).
// mode 0: bf16 row-major C   mode 1: V-transposed bf16 [b,h,d,s]
// mode 2: f32 row-major C
// ---------------------------------------------------------------------------
__global__ __launch_bounds__(128) void gemm_bt_kernel(
    const unsigned short* __restrict__ A,
    const unsigned short* __restrict__ Bt,
    void* __restrict__ Cout,
    int M, int N, int Kd, int mode) {
  const int lane  = threadIdx.x & 31;
  const int wave  = threadIdx.x >> 5;
  const int col16 = lane & 15;
  const int n0 = blockIdx.x * 64;
  const int m0 = blockIdx.y * 128 + wave * 32;

  const unsigned short* aRow0 = A + (size_t)(m0 + col16) * Kd;
  const unsigned short* aRow1 = A + (size_t)(m0 + 16 + col16) * Kd;

  v8f acc[2][4];
#pragma unroll
  for (int s2 = 0; s2 < 2; ++s2)
#pragma unroll
    for (int t = 0; t < 4; ++t) acc[s2][t] = {};

#pragma unroll 4
  for (int kk = 0; kk < Kd; kk += 32) {
    v16bf a0 = load_afrag(aRow0 + kk, lane);
    v16bf a1 = load_afrag(aRow1 + kk, lane);
#pragma unroll
    for (int t = 0; t < 4; ++t) {
      const int col = n0 + t * 16 + col16;
      v16bf bfr = load_bfrag(Bt + (size_t)col * Kd + kk, lane);
      acc[0][t] = wmma_bf16(a0, bfr, acc[0][t]);
      acc[1][t] = wmma_bf16(a1, bfr, acc[1][t]);
    }
  }

  const int rbase = (lane < 16) ? 0 : 8;   // C layout: M = r + 8*half
  if (mode == 2) {
    float* C = (float*)Cout;
#pragma unroll
    for (int s2 = 0; s2 < 2; ++s2)
#pragma unroll
      for (int t = 0; t < 4; ++t) {
        int col = n0 + t * 16 + col16;
#pragma unroll
        for (int r = 0; r < 8; ++r)
          C[(size_t)(m0 + s2 * 16 + rbase + r) * N + col] = acc[s2][t][r];
      }
  } else if (mode == 1) {
    unsigned short* C = (unsigned short*)Cout;
#pragma unroll
    for (int s2 = 0; s2 < 2; ++s2)
#pragma unroll
      for (int t = 0; t < 4; ++t) {
        int col = n0 + t * 16 + col16;
        int h = col >> 6, d = col & 63;
#pragma unroll
        for (int r = 0; r < 8; ++r) {
          int m = m0 + s2 * 16 + rbase + r;
          int b = m >> 11, s = m & 2047;
          C[((size_t)((b * N_HEADS + h) * HD + d) << 11) + s] = f2bf(acc[s2][t][r]);
        }
      }
  } else {
    unsigned short* C = (unsigned short*)Cout;
#pragma unroll
    for (int s2 = 0; s2 < 2; ++s2)
#pragma unroll
      for (int t = 0; t < 4; ++t) {
        int col = n0 + t * 16 + col16;
#pragma unroll
        for (int r = 0; r < 8; ++r)
          C[(size_t)(m0 + s2 * 16 + rbase + r) * N + col] = f2bf(acc[s2][t][r]);
      }
  }
}

// ---------------------------------------------------------------------------
// Flash attention (causal). Grid = (S/64, H, B); block = 128 (4 waves).
// Each wave owns 16 query rows; 64-key tiles iterated up to the diagonal.
// ---------------------------------------------------------------------------
__global__ __launch_bounds__(128) void flash_attn_kernel(
    const unsigned short* __restrict__ Q,    // [4096][1024] bf16
    const unsigned short* __restrict__ Kb,   // [4096][1024] bf16
    const unsigned short* __restrict__ Vt,   // [B*H*64][2048] bf16 (d-major)
    unsigned short* __restrict__ O) {        // [4096][1024] bf16
  __shared__ unsigned short Plds[4][16 * 64];   // per-wave P staging (8 KB)

  const int lane  = threadIdx.x & 31;
  const int wave  = threadIdx.x >> 5;
  const int col16 = lane & 15;
  const int half  = (lane < 16) ? 0 : 1;
  const int qt = blockIdx.x;   // query tile 0..31
  const int h  = blockIdx.y;
  const int b  = blockIdx.z;

  // Q fragments for this wave's 16 rows (Kd = 64 -> two 16x32 A-fragments)
  const int qrow = qt * 64 + wave * 16 + col16;
  const unsigned short* qPtr =
      Q + (size_t)(b * S_LEN + qrow) * EMB + h * HD;
  v16bf aQ[2];
  aQ[0] = load_afrag(qPtr, lane);
  aQ[1] = load_afrag(qPtr + 32, lane);

  float mrow[8], lrow[8];
  v8f oacc[4];
#pragma unroll
  for (int r = 0; r < 8; ++r) { mrow[r] = -1e30f; lrow[r] = 0.0f; }
#pragma unroll
  for (int t = 0; t < 4; ++t) oacc[t] = {};

  unsigned short* myP = &Plds[wave][0];

  for (int kt = 0; kt <= qt; ++kt) {
    // Prefetch next key/value tiles into cache (global_prefetch_b8)
    if (kt + 1 <= qt) {
      const unsigned short* nk =
          Kb + (size_t)(b * S_LEN + (kt + 1) * 64 + lane) * EMB + h * HD;
      __builtin_prefetch(nk, 0, 0);
      const unsigned short* nv =
          Vt + (size_t)((b * N_HEADS + h) * HD + lane) * S_LEN + (kt + 1) * 64;
      __builtin_prefetch(nv, 0, 0);
    }

    // logits S = Q * K^T : K rows are contiguous -> direct Bt fragments
    v8f sfr[4];
#pragma unroll
    for (int t = 0; t < 4; ++t) {
      const int kcol = kt * 64 + t * 16 + col16;
      const unsigned short* kPtr =
          Kb + (size_t)(b * S_LEN + kcol) * EMB + h * HD;
      v8f c = {};
      c = wmma_bf16(aQ[0], load_bfrag(kPtr, lane), c);
      c = wmma_bf16(aQ[1], load_bfrag(kPtr + 32, lane), c);
      sfr[t] = c;
    }

    // scale by 1/sqrt(64) and apply causal mask on the diagonal tile
#pragma unroll
    for (int t = 0; t < 4; ++t) {
#pragma unroll
      for (int r = 0; r < 8; ++r) {
        float v = sfr[t][r] * 0.125f;
        if (kt == qt) {
          int qpos = wave * 16 + half * 8 + r;
          int kpos = t * 16 + col16;
          if (qpos < kpos) v = -1e30f;
        }
        sfr[t][r] = v;
      }
    }

    // online softmax: row max across the 16-lane half-group
    float mnew[8];
#pragma unroll
    for (int r = 0; r < 8; ++r) {
      float v = mrow[r];
#pragma unroll
      for (int t = 0; t < 4; ++t) v = fmaxf(v, sfr[t][r]);
#pragma unroll
      for (int off = 1; off < 16; off <<= 1)
        v = fmaxf(v, __shfl_xor(v, off, 32));
      mnew[r] = v;
    }

    float alpha[8], rsum[8];
#pragma unroll
    for (int r = 0; r < 8; ++r) {
      alpha[r] = __expf(mrow[r] - mnew[r]);
      rsum[r] = 0.0f;
    }

    // P = exp(S - mnew): accumulate row sums, stage bf16 P into per-wave LDS
#pragma unroll
    for (int t = 0; t < 4; ++t) {
#pragma unroll
      for (int r = 0; r < 8; ++r) {
        float p = __expf(sfr[t][r] - mnew[r]);
        rsum[r] += p;
        myP[(half * 8 + r) * 64 + t * 16 + col16] = f2bf(p);
      }
    }

#pragma unroll
    for (int r = 0; r < 8; ++r) {
      float v = rsum[r];
#pragma unroll
      for (int off = 1; off < 16; off <<= 1)
        v += __shfl_xor(v, off, 32);
      lrow[r] = lrow[r] * alpha[r] + v;
      mrow[r] = mnew[r];
    }

    // rescale running output accumulators
#pragma unroll
    for (int t = 0; t < 4; ++t)
#pragma unroll
      for (int r = 0; r < 8; ++r) oacc[t][r] = oacc[t][r] * alpha[r];

    // per-wave LDS: writes then reads by same wave -> wait DS counter
    asm volatile("s_wait_dscnt 0" ::: "memory");

    // Reload P (16x64) from LDS in A-fragment layout; accumulate P * V
    v16bf aP[2];
    aP[0] = load_afrag(myP + col16 * 64, lane);
    aP[1] = load_afrag(myP + col16 * 64 + 32, lane);

#pragma unroll
    for (int t = 0; t < 4; ++t) {
      const int d = t * 16 + col16;
      const unsigned short* vPtr =
          Vt + (size_t)((b * N_HEADS + h) * HD + d) * S_LEN + kt * 64;
      oacc[t] = wmma_bf16(aP[0], load_bfrag(vPtr, lane), oacc[t]);
      oacc[t] = wmma_bf16(aP[1], load_bfrag(vPtr + 32, lane), oacc[t]);
    }
  }

  // normalize and write O (bf16, [m][h*64+d])
  float inv[8];
#pragma unroll
  for (int r = 0; r < 8; ++r) inv[r] = 1.0f / lrow[r];
  const int rbase = half * 8;
#pragma unroll
  for (int t = 0; t < 4; ++t) {
    int d = t * 16 + col16;
#pragma unroll
    for (int r = 0; r < 8; ++r) {
      int m = b * S_LEN + qt * 64 + wave * 16 + rbase + r;
      O[(size_t)m * EMB + h * HD + d] = f2bf(oacc[t][r] * inv[r]);
    }
  }
}

// ---------------------------------------------------------------------------
// Host-side launcher
// ---------------------------------------------------------------------------
extern "C" void kernel_launch(void* const* d_in, const int* in_sizes, int n_in,
                              void* d_out, int out_size, void* d_ws, size_t ws_size,
                              hipStream_t stream) {
  (void)in_sizes; (void)n_in; (void)out_size; (void)ws_size;

  const float* x  = (const float*)d_in[0];
  const float* wq = (const float*)d_in[1];
  const float* wk = (const float*)d_in[2];
  const float* wv = (const float*)d_in[3];
  const float* wo = (const float*)d_in[4];

  char* ws = (char*)d_ws;
  const size_t MB = 1024u * 1024u;
  unsigned short* Xb  = (unsigned short*)(ws + 0 * MB);    // 8 MB  [4096][1024]
  unsigned short* Wqt = (unsigned short*)(ws + 8 * MB);    // 2 MB  [1024][1024] (transposed)
  unsigned short* Wkt = (unsigned short*)(ws + 10 * MB);   // 2 MB
  unsigned short* Wvt = (unsigned short*)(ws + 12 * MB);   // 2 MB
  unsigned short* Wob = (unsigned short*)(ws + 14 * MB);   // 2 MB  [E][HK] (already B^T form)
  unsigned short* Qb  = (unsigned short*)(ws + 16 * MB);   // 8 MB
  unsigned short* Kb  = (unsigned short*)(ws + 24 * MB);   // 8 MB
  unsigned short* Vt  = (unsigned short*)(ws + 32 * MB);   // 8 MB  [B*H*64][2048]
  unsigned short* Ob  = (unsigned short*)(ws + 40 * MB);   // 8 MB

  // 1) conversions
  convert_bf16_kernel<<<(MROWS * EMB) / 256, 256, 0, stream>>>(x, Xb, MROWS * EMB);
  transpose_convert_kernel<<<(EMB * EMB) / 256, 256, 0, stream>>>(wq, Wqt);
  transpose_convert_kernel<<<(EMB * EMB) / 256, 256, 0, stream>>>(wk, Wkt);
  transpose_convert_kernel<<<(EMB * EMB) / 256, 256, 0, stream>>>(wv, Wvt);
  convert_bf16_kernel<<<(EMB * EMB) / 256, 256, 0, stream>>>(wo, Wob, EMB * EMB);

  // 2) QKV projections (32x64 per wave, 128x64 per block)
  dim3 gg(EMB / 64, MROWS / 128);
  gemm_bt_kernel<<<gg, 128, 0, stream>>>(Xb, Wqt, Qb, MROWS, EMB, EMB, 0);
  gemm_bt_kernel<<<gg, 128, 0, stream>>>(Xb, Wkt, Kb, MROWS, EMB, EMB, 0);
  gemm_bt_kernel<<<gg, 128, 0, stream>>>(Xb, Wvt, Vt, MROWS, EMB, EMB, 1);

  // 3) causal flash attention
  dim3 gf(S_LEN / 64, N_HEADS, 2);
  flash_attn_kernel<<<gf, 128, 0, stream>>>(Qb, Kb, Vt, Ob);

  // 4) output projection (f32 result)
  gemm_bt_kernel<<<gg, 128, 0, stream>>>(Ob, Wob, d_out, MROWS, EMB, EMB, 2);
}
